// PatchFilter_29781303231202
// MI455X (gfx1250) — compile-verified, matrix operand order checked by reading, not run on hardware
//
#include <hip/hip_runtime.h>
#include <hip/hip_bf16.h>

// ---------------------------------------------------------------------------
// PatchFilter: cosine-sim top-k attention mask for MI455X (gfx1250, wave32)
//   tokens [8,4096,64] f32  ->  attn_mask [8,1,4096,4096] f32 {0, -inf}
// Strategy:
//   K1 normalize rows -> f16 workspace (WMMA operand precision)
//   K2 fused WMMA GEMM (sim stripes) + per-row top-8 (never materialize sim)
//   K3 analytic fill of seasonal band mask with NT float4 stores (512MB, BW bound)
//   K4 scatter 0.0 at top-8 indices
// ---------------------------------------------------------------------------

typedef _Float16 v16h __attribute__((ext_vector_type(16)));
typedef _Float16 v8h  __attribute__((ext_vector_type(8)));
typedef _Float16 v2h  __attribute__((ext_vector_type(2)));
typedef float    v8f  __attribute__((ext_vector_type(8)));
typedef float    v4f  __attribute__((ext_vector_type(4)));

#define BATCH 8
#define LSEQ  4096
#define DDIM  64
#define TOPK  8
#define CHUNK 256            // sim columns buffered in LDS per top-k pass
#define LDS_STRIDE 260       // 260 % 64 == 4 -> 16 scanning lanes hit distinct banks

// ---------------- Kernel 1: L2 row normalize, f32 -> f16 -------------------
__global__ __launch_bounds__(256) void pf_normalize(const float* __restrict__ tokens,
                                                    _Float16* __restrict__ xh) {
  const int lane = threadIdx.x & 31;
  const int wave = threadIdx.x >> 5;
  const int row  = blockIdx.x * 8 + wave;          // 32768 rows total
  const float2 t = ((const float2*)(tokens + (size_t)row * DDIM))[lane];
  float s = t.x * t.x + t.y * t.y;
  s += __shfl_xor(s, 16);
  s += __shfl_xor(s, 8);
  s += __shfl_xor(s, 4);
  s += __shfl_xor(s, 2);
  s += __shfl_xor(s, 1);
  const float inv = 1.0f / fmaxf(sqrtf(s), 1e-12f);
  v2h o;
  o[0] = (_Float16)(t.x * inv);
  o[1] = (_Float16)(t.y * inv);
  ((v2h*)xh)[(size_t)row * 32 + lane] = o;
}

// ------- Kernel 2: fused WMMA similarity + register-resident top-8 ---------
// One wave per 16-row stripe. grid = 8 * (4096/16) = 2048 blocks of 32.
__global__ __launch_bounds__(32) void pf_sim_topk(const _Float16* __restrict__ xh,
                                                  int* __restrict__ topk) {
  __shared__ float tile[16][LDS_STRIDE];

  const int lane  = threadIdx.x;          // 0..31
  const int b     = blockIdx.x >> 8;
  const int itile = blockIdx.x & 255;
  const int half  = lane >> 4;            // 0: K{0-7,16-23}, 1: K{8-15,24-31}
  const int r     = lane & 15;            // A row / B col / C col within tile
  const int ko    = half * 8;

  const _Float16* xb = xh + (size_t)b * LSEQ * DDIM;

  // A tile: rows itile*16 + r, split into two K=32 operands per WMMA layout
  const _Float16* arow = xb + (size_t)(itile * 16 + r) * DDIM;
  v8h a0lo = *(const v8h*)(arow + 0  + ko);
  v8h a0hi = *(const v8h*)(arow + 16 + ko);
  v8h a1lo = *(const v8h*)(arow + 32 + ko);
  v8h a1hi = *(const v8h*)(arow + 48 + ko);
  v16h A0 = __builtin_shufflevector(a0lo, a0hi, 0,1,2,3,4,5,6,7,8,9,10,11,12,13,14,15);
  v16h A1 = __builtin_shufflevector(a1lo, a1hi, 0,1,2,3,4,5,6,7,8,9,10,11,12,13,14,15);

  // per-lane top-8 state (lane m owns sim row m of this stripe)
  float vals[TOPK];
  int   idxs[TOPK];
#pragma unroll
  for (int s = 0; s < TOPK; ++s) { vals[s] = -__builtin_inff(); idxs[s] = 0; }
  float minv   = -__builtin_inff();
  int   minslot = 0;

  for (int chunk = 0; chunk < LSEQ / CHUNK; ++chunk) {
#pragma unroll 1
    for (int jt = 0; jt < CHUNK / 16; ++jt) {
      const int j0 = chunk * CHUNK + jt * 16;
      const _Float16* brow = xb + (size_t)(j0 + r) * DDIM;  // B = x^T tile == rows of x
      __builtin_prefetch(brow + DDIM * 16, 0, 1);           // next j-tile -> global_prefetch
      v8h b0lo = *(const v8h*)(brow + 0  + ko);
      v8h b0hi = *(const v8h*)(brow + 16 + ko);
      v8h b1lo = *(const v8h*)(brow + 32 + ko);
      v8h b1hi = *(const v8h*)(brow + 48 + ko);
      v16h B0 = __builtin_shufflevector(b0lo, b0hi, 0,1,2,3,4,5,6,7,8,9,10,11,12,13,14,15);
      v16h B1 = __builtin_shufflevector(b1lo, b1hi, 0,1,2,3,4,5,6,7,8,9,10,11,12,13,14,15);

      v8f c = {};
      c = __builtin_amdgcn_wmma_f32_16x16x32_f16(false, A0, false, B0, (short)0, c, false, false);
      c = __builtin_amdgcn_wmma_f32_16x16x32_f16(false, A1, false, B1, (short)0, c, false, false);

      // C layout: VGPR v, lanes0-15 -> (M=v, N=lane); lanes16-31 -> (M=8+v, N=lane-16)
#pragma unroll
      for (int v = 0; v < 8; ++v)
        tile[v + 8 * half][jt * 16 + r] = c[v];
    }
    __syncthreads();

    if (lane < 16) {
      const int m = lane;
#pragma unroll 1
      for (int c4 = 0; c4 < CHUNK / 4; ++c4) {
        const v4f vv = *(const v4f*)&tile[m][c4 * 4];   // ds_load_b128, conflict-free
        const int jb = chunk * CHUNK + c4 * 4;
#pragma unroll
        for (int e = 0; e < 4; ++e) {
          const float v = vv[e];
          if (v > minv) {                               // rare slow path
#pragma unroll
            for (int s = 0; s < TOPK; ++s)
              if (s == minslot) { vals[s] = v; idxs[s] = jb + e; }
            minv = vals[0]; minslot = 0;
#pragma unroll
            for (int s = 1; s < TOPK; ++s)
              if (vals[s] < minv) { minv = vals[s]; minslot = s; }
          }
        }
      }
    }
    __syncthreads();
  }

  if (lane < 16) {
    const int row = itile * 16 + lane;
    int* dst = topk + ((size_t)b * LSEQ + row) * TOPK;
#pragma unroll
    for (int s = 0; s < TOPK; ++s) dst[s] = idxs[s];
  }
}

// ---- Kernel 3: analytic seasonal-band fill, NT float4 stores (512 MB) -----
__global__ __launch_bounds__(256) void pf_fill(float* __restrict__ out) {
  const size_t t = (size_t)blockIdx.x * blockDim.x + threadIdx.x; // 33,554,432 threads
  const int j0 = (int)(t & 1023) * 4;
  const int i  = (int)(t >> 10) & 4095;
  const float NEG_INF = -__builtin_inff();
  v4f v;
#pragma unroll
  for (int e = 0; e < 4; ++e) {
    const int d = i - (j0 + e);
    v[e] = (d == 0 || d == 24 || d == -24 || d == 48 || d == -48) ? 0.0f : NEG_INF;
  }
  __builtin_nontemporal_store(v, (v4f*)(out + t * 4));
}

// ---------------- Kernel 4: scatter 0.0 at top-k positions -----------------
__global__ __launch_bounds__(256) void pf_scatter(const int* __restrict__ topk,
                                                  float* __restrict__ out) {
  const int t = blockIdx.x * blockDim.x + threadIdx.x;   // 262144
  const int j = topk[t];
  const int row = t >> 3;                                 // b*L + i
  out[(size_t)row * LSEQ + j] = 0.0f;
}

// ---------------------------------------------------------------------------
extern "C" void kernel_launch(void* const* d_in, const int* in_sizes, int n_in,
                              void* d_out, int out_size, void* d_ws, size_t ws_size,
                              hipStream_t stream) {
  const float* tokens = (const float*)d_in[0];
  float* out = (float*)d_out;

  _Float16* xh  = (_Float16*)d_ws;                                  // 4 MB
  int* topk = (int*)((char*)d_ws + (size_t)BATCH * LSEQ * DDIM * sizeof(_Float16));

  pf_normalize<<<(BATCH * LSEQ) / 8, 256, 0, stream>>>(tokens, xh);
  pf_sim_topk<<<BATCH * (LSEQ / 16), 32, 0, stream>>>(xh, topk);
  pf_fill<<<(size_t)BATCH * LSEQ * (LSEQ / 4) / 256, 256, 0, stream>>>(out);
  pf_scatter<<<(BATCH * LSEQ * TOPK) / 256, 256, 0, stream>>>(topk, out);
}
